// MoE_23046794510696
// MI455X (gfx1250) — matrix operand order repaired and checked by reading, not compiled, not run on hardware
//
#include <hip/hip_runtime.h>
#include <hip/hip_bf16.h>
#include <math.h>

// ---------------------------------------------------------------------------
// MoE (top-2 of 8, 7 MLP experts + identity), N=16384 D=512 H=2048.
// Compute-bound (AI ~3800 flop/byte) -> bf16 WMMA 16x16x32, f32 accumulate.
// Weights pre-converted once to bf16 in a fragment-major swizzled layout in
// workspace; hot kernel stages W slices into LDS with the Tensor Data Mover
// (tensor_load_to_lds + s_wait_tensorcnt), h tile stays in LDS end to end.
// ---------------------------------------------------------------------------

#define MOE_D 512
#define MOE_H 2048
#define MOE_E 8
#define MOE_EPS 2.220446049250313e-16f

typedef __attribute__((ext_vector_type(16))) __bf16        v16bf;
typedef __attribute__((ext_vector_type(8)))  float         v8f;
typedef __attribute__((ext_vector_type(4)))  unsigned int  v4u;
typedef __attribute__((ext_vector_type(8)))  int           v8i;
typedef __attribute__((ext_vector_type(4)))  int           v4i;

union BF16x16 {
    v16bf v;
    unsigned short u[16];
    unsigned int   w[8];
};

__device__ __forceinline__ unsigned short f2bf(float f) {
    // float -> bf16 bits, round-to-nearest-even (inputs finite)
    unsigned int x = __float_as_uint(f);
    unsigned int r = x + 0x7FFFu + ((x >> 16) & 1u);
    return (unsigned short)(r >> 16);
}

__device__ __forceinline__ v8f vzero8() {
    v8f z;
#pragma unroll
    for (int i = 0; i < 8; ++i) z[i] = 0.0f;
    return z;
}

// A fragment: 16x32 bf16 (ISA 7.12.2): lanes 0-15 -> M=lane, K={0..7,16..23};
// lanes 16-31 -> M=lane-16, K={8..15,24..31}. Two 16B LDS vector reads.
__device__ __forceinline__ v16bf load_a_frag(const unsigned short* base, int stride,
                                             int k0, int lane) {
    const int m     = lane & 15;
    const int kbase = (lane >> 4) * 8;
    const unsigned short* p = base + m * stride + k0 + kbase;
    uint4 lo = *(const uint4*)p;          // K kbase+0..7
    uint4 hi = *(const uint4*)(p + 16);   // K kbase+16..23
    BF16x16 A;
    A.w[0] = lo.x; A.w[1] = lo.y; A.w[2] = lo.z; A.w[3] = lo.w;
    A.w[4] = hi.x; A.w[5] = hi.y; A.w[6] = hi.z; A.w[7] = hi.w;
    return A.v;
}

// B fragment: 32x16 bf16 (KxN). wbuf is fragment-major: wbuf[col*32 + k],
// so lane (n=l&15, kb=(l>>4)*8) reads K kb+0..7 and kb+16..23 = two b128 LDS loads.
__device__ __forceinline__ v16bf load_b_frag(const unsigned short* wbuf,
                                             int col0, int lane) {
    const int n  = lane & 15;
    const int kb = (lane >> 4) * 8;
    const unsigned short* p = wbuf + (col0 + n) * 32 + kb;
    uint4 lo = *(const uint4*)p;
    uint4 hi = *(const uint4*)(p + 16);
    BF16x16 B;
    B.w[0] = lo.x; B.w[1] = lo.y; B.w[2] = lo.z; B.w[3] = lo.w;
    B.w[4] = hi.x; B.w[5] = hi.y; B.w[6] = hi.z; B.w[7] = hi.w;
    return B.v;
}

// 1-D contiguous global(bf16) -> LDS copy via Tensor Data Mover.
// D# per ISA 8.3-8.5: count=1, type=2; data_size=1 (2 bytes);
// tensor_dim0 = tile_dim0 = nelem; tensor_dim1 = tile_dim1 = 1.
// This toolchain exposes the 6-arg builtin: (v4u, v8i, v4i, v4i, v8i, i32).
__device__ __forceinline__ void tdm_load_1d(const void* gsrc, unsigned lds_off,
                                            unsigned nelem) {
    const unsigned long long ga = (unsigned long long)(size_t)gsrc;
    v4u g0;
    g0[0] = 1u;                                  // count=1, user mode
    g0[1] = lds_off;                             // lds_addr (bytes)
    g0[2] = (unsigned)(ga & 0xFFFFFFFFu);        // global_addr[31:0]
    g0[3] = (unsigned)(ga >> 32) | (2u << 30);   // global_addr[56:32] | type=2
    v8i g1;
    g1[0] = 0x00010000;                          // data_size=1 (2B), wg_mask=0
    g1[1] = (int)((nelem & 0xFFFFu) << 16);      // tensor_dim0[15:0]
    g1[2] = (int)(((nelem >> 16) & 0xFFFFu) | (1u << 16)); // dim0 hi | tensor_dim1=1
    g1[3] = (int)(nelem << 16);                  // tile_dim0
    g1[4] = 1;                                   // tile_dim1=1
    g1[5] = (int)nelem;                          // tensor_dim0_stride[31:0]
    g1[6] = 0;
    g1[7] = 0;
    v4i z4; z4[0] = 0; z4[1] = 0; z4[2] = 0; z4[3] = 0;
    v8i z8;
#pragma unroll
    for (int i = 0; i < 8; ++i) z8[i] = 0;
    __builtin_amdgcn_tensor_load_to_lds(g0, g1, z4, z4, z8, 0);
}

// ---------------------------------------------------------------------------
// Kernel 0a/0b: one-time fp32 -> bf16 weight conversion into swizzled layout.
// W1swz[e][hcI][k0I][cc][kk] (hcI:8 chunks of 256 cols, k0I:16 slices of 32 k)
// W2swz[e][hcI][k0I][cc][kk] (hcI:8 k-chunks of 256, k0I:8 slices, cc:512 cols)
// ---------------------------------------------------------------------------
__global__ __launch_bounds__(256)
void conv_w1_kernel(const float* __restrict__ W1, unsigned short* __restrict__ W1swz) {
    const int total = (MOE_E - 1) * MOE_D * MOE_H;
    for (int d = blockIdx.x * 256 + threadIdx.x; d < total; d += gridDim.x * 256) {
        const int kk  = d & 31;
        const int cc  = (d >> 5) & 255;
        const int k0I = (d >> 13) & 15;
        const int hcI = (d >> 17) & 7;
        const int e   = d >> 20;
        W1swz[d] = f2bf(W1[((size_t)(e * MOE_D + k0I * 32 + kk)) * MOE_H + hcI * 256 + cc]);
    }
}

__global__ __launch_bounds__(256)
void conv_w2_kernel(const float* __restrict__ W2, unsigned short* __restrict__ W2swz) {
    const int total = (MOE_E - 1) * MOE_H * MOE_D;
    for (int d = blockIdx.x * 256 + threadIdx.x; d < total; d += gridDim.x * 256) {
        const int kk  = d & 31;
        const int cc  = (d >> 5) & 511;
        const int k0I = (d >> 14) & 7;
        const int hcI = (d >> 17) & 7;
        const int e   = d >> 20;
        W2swz[d] = f2bf(W2[((size_t)(e * MOE_H + hcI * 256 + k0I * 32 + kk)) * MOE_D + cc]);
    }
}

// ---------------------------------------------------------------------------
// Kernel 1: gating. One wave per row: logits = x_row @ w_gate, top-2, softmax.
// ---------------------------------------------------------------------------
__global__ __launch_bounds__(256)
void moe_gate_kernel(const float* __restrict__ x, const float* __restrict__ wg,
                     float* __restrict__ gates) {
    const int lane = threadIdx.x & 31;
    const int row  = blockIdx.x * 8 + (threadIdx.x >> 5);

    float p[MOE_E];
#pragma unroll
    for (int e = 0; e < MOE_E; ++e) p[e] = 0.0f;

    const float* xr = x + (size_t)row * MOE_D;
    for (int d = lane; d < MOE_D; d += 32) {
        const float xv = xr[d];
        const float* wr = wg + (size_t)d * MOE_E;
#pragma unroll
        for (int e = 0; e < MOE_E; ++e) p[e] += xv * wr[e];
    }
#pragma unroll
    for (int off = 16; off > 0; off >>= 1) {
#pragma unroll
        for (int e = 0; e < MOE_E; ++e) p[e] += __shfl_xor(p[e], off, 32);
    }

    if (lane == 0) {
        int i1 = 0; float v1 = p[0];
#pragma unroll
        for (int e = 1; e < MOE_E; ++e) { if (p[e] > v1) { v1 = p[e]; i1 = e; } }
        int i2 = -1; float v2 = 0.0f;
#pragma unroll
        for (int e = 0; e < MOE_E; ++e) {
            if (e == i1) continue;
            if (i2 < 0 || p[e] > v2) { v2 = p[e]; i2 = e; }
        }
        const float g1 = 1.0f / (1.0f + __expf(v2 - v1));  // softmax over top-2
        const float g2 = 1.0f - g1;
        float* gr = gates + (size_t)row * MOE_E;
#pragma unroll
        for (int e = 0; e < MOE_E; ++e)
            gr[e] = (e == i1) ? g1 : (e == i2) ? g2 : 0.0f;
    }
}

// ---------------------------------------------------------------------------
// Kernel 2: fused expert MLP. Block = 256 threads (8 waves), 16-row token tile.
// ---------------------------------------------------------------------------
__global__ __launch_bounds__(256)
void moe_fused_kernel(const float* __restrict__ x,
                      const unsigned short* __restrict__ W1swz,
                      const float* __restrict__ b1,
                      const unsigned short* __restrict__ W2swz,
                      const float* __restrict__ b2,
                      const float* __restrict__ gates,
                      float* __restrict__ out) {
    __shared__ __align__(16) unsigned short xs[16 * 520];    // x tile, bf16
    __shared__ __align__(16) unsigned short hs[16 * 264];    // h chunk, bf16
    __shared__ __align__(16) unsigned short wbuf[512 * 32];  // W slice, frag-major
    __shared__ float gbuf[16];
    __shared__ float g7buf[16];

    const int tid  = threadIdx.x;
    const int lane = tid & 31;
    const int wave = tid >> 5;          // 0..7
    const int row0 = blockIdx.x * 16;
    const unsigned wbuf_lds = (unsigned)(size_t)&wbuf[0];   // LDS byte offset

    // Stage x tile fp32 -> bf16 (coalesced).
    for (int j = tid; j < 16 * MOE_D; j += 256) {
        const int r = j >> 9, c = j & 511;
        xs[r * 520 + c] = f2bf(x[(size_t)(row0 + r) * MOE_D + c]);
    }
    if (tid < 16) g7buf[tid] = gates[(size_t)(row0 + tid) * MOE_E + (MOE_E - 1)];
    __syncthreads();

    v8f acc[4];
#pragma unroll
    for (int t = 0; t < 4; ++t) acc[t] = vzero8();

    for (int e = 0; e < MOE_E - 1; ++e) {
        __syncthreads();
        if (tid < 16) gbuf[tid] = gates[(size_t)(row0 + tid) * MOE_E + e];
        __syncthreads();

        float gsum = 0.0f;
#pragma unroll
        for (int t = 0; t < 16; ++t) gsum += fabsf(gbuf[t]);
        if (gsum == 0.0f) continue;   // uniform skip: expert unused in this tile

        const unsigned short* W1e = W1swz + (size_t)e * MOE_D * MOE_H;
        const unsigned short* W2e = W2swz + (size_t)e * MOE_H * MOE_D;
        const float* b1e = b1 + (size_t)e * MOE_H;
        const float* b2e = b2 + (size_t)e * MOE_D;

        v8f c2[4];
#pragma unroll
        for (int t = 0; t < 4; ++t) c2[t] = vzero8();

        for (int hc = 0; hc < MOE_H; hc += 256) {
            const int hcI = hc >> 8;
            // ---------------- GEMM1: x_tile @ W1[:, hc:hc+256] ----------------
            v8f c1[2];
#pragma unroll
            for (int t = 0; t < 2; ++t) c1[t] = vzero8();

            for (int k0 = 0; k0 < MOE_D; k0 += 32) {
                __syncthreads();   // previous wbuf consumers done
                const unsigned short* slice =
                    W1e + ((size_t)(hcI * 16 + (k0 >> 5))) * (256 * 32);
                if (wave == 0) {
                    if (lane == 0 && k0 + 32 < MOE_D)
                        __builtin_prefetch(slice + 256 * 32, 0, 1);
                    tdm_load_1d(slice, wbuf_lds, 256 * 32);     // 16 KB DMA
                    __builtin_amdgcn_s_wait_tensorcnt(0);
                }
                __syncthreads();   // wbuf ready

                const v16bf a = load_a_frag(xs, 520, k0, lane);
#pragma unroll
                for (int t = 0; t < 2; ++t) {
                    const v16bf b = load_b_frag(wbuf, wave * 32 + t * 16, lane);
                    c1[t] = __builtin_amdgcn_wmma_f32_16x16x32_bf16(
                        false, a, false, b, (short)0, c1[t], false, false);
                }
            }

            // bias + relu, spill h chunk to LDS as bf16
#pragma unroll
            for (int t = 0; t < 2; ++t) {
                const int cl = wave * 32 + t * 16 + (lane & 15);
                const float bb = b1e[hc + cl];
#pragma unroll
                for (int r = 0; r < 8; ++r) {
                    float v = c1[t][r] + bb;
                    v = v > 0.0f ? v : 0.0f;
                    hs[(((lane >> 4) << 3) + r) * 264 + cl] = f2bf(v);
                }
            }
            __syncthreads();   // h chunk visible

            // ---------------- GEMM2: h_chunk @ W2[hc:hc+256, :] ----------------
            for (int k0 = 0; k0 < 256; k0 += 32) {
                __syncthreads();   // previous wbuf consumers done
                const unsigned short* slice =
                    W2e + ((size_t)(hcI * 8 + (k0 >> 5))) * (512 * 32);
                if (wave == 0) {
                    tdm_load_1d(slice, wbuf_lds, 512 * 32);     // 32 KB DMA
                    __builtin_amdgcn_s_wait_tensorcnt(0);
                }
                __syncthreads();   // wbuf ready

                const v16bf a2 = load_a_frag(hs, 264, k0, lane);
#pragma unroll
                for (int t = 0; t < 4; ++t) {
                    const v16bf b2f = load_b_frag(wbuf, wave * 64 + t * 16, lane);
                    c2[t] = __builtin_amdgcn_wmma_f32_16x16x32_bf16(
                        false, a2, false, b2f, (short)0, c2[t], false, false);
                }
            }
        }

        // acc += gate_row * (c2 + b2)
#pragma unroll
        for (int t = 0; t < 4; ++t) {
            const int col = wave * 64 + t * 16 + (lane & 15);
            const float bb2 = b2e[col];
#pragma unroll
            for (int r = 0; r < 8; ++r) {
                const int lr = ((lane >> 4) << 3) + r;
                acc[t][r] += gbuf[lr] * (c2[t][r] + bb2);
            }
        }
    }

    // identity expert (fp32 x) + zero->EPS + store
#pragma unroll
    for (int t = 0; t < 4; ++t) {
        const int col = wave * 64 + t * 16 + (lane & 15);
#pragma unroll
        for (int r = 0; r < 8; ++r) {
            const int lr  = ((lane >> 4) << 3) + r;
            const int row = row0 + lr;
            float v = acc[t][r] + g7buf[lr] * x[(size_t)row * MOE_D + col];
            if (v == 0.0f) v = MOE_EPS;
            out[(size_t)row * MOE_D + col] = v;
        }
    }
}

// ---------------------------------------------------------------------------
extern "C" void kernel_launch(void* const* d_in, const int* in_sizes, int n_in,
                              void* d_out, int out_size, void* d_ws, size_t ws_size,
                              hipStream_t stream) {
    const float* x  = (const float*)d_in[0];
    const float* wg = (const float*)d_in[1];
    const float* W1 = (const float*)d_in[2];
    const float* b1 = (const float*)d_in[3];
    const float* W2 = (const float*)d_in[4];
    const float* b2 = (const float*)d_in[5];
    // d_in[6] = k (always 2 in this problem)
    float* out = (float*)d_out;

    const int N = in_sizes[0] / MOE_D;   // 16384
    const size_t nW = (size_t)(MOE_E - 1) * MOE_D * MOE_H;   // 7,340,032

    // ws layout: gates [N*8 f32] | W1swz [nW bf16] | W2swz [nW bf16]  (~29.9 MB)
    float*          gates = (float*)d_ws;
    unsigned short* W1swz = (unsigned short*)((char*)d_ws + (size_t)N * MOE_E * 4);
    unsigned short* W2swz = W1swz + nW;

    conv_w1_kernel<<<dim3(2048), dim3(256), 0, stream>>>(W1, W1swz);
    conv_w2_kernel<<<dim3(2048), dim3(256), 0, stream>>>(W2, W2swz);
    moe_gate_kernel<<<dim3(N / 8), dim3(256), 0, stream>>>(x, wg, gates);
    moe_fused_kernel<<<dim3(N / 16), dim3(256), 0, stream>>>(x, W1swz, b1, W2swz,
                                                             b2, gates, out);
}
